// SpikingLSTM_35089882808558
// MI455X (gfx1250) — matrix operand order repaired and checked by reading, not compiled, not run on hardware
//
#include <hip/hip_runtime.h>
#include <hip/hip_bf16.h>

typedef __attribute__((ext_vector_type(16))) __bf16 v16bf;
typedef __attribute__((ext_vector_type(8)))  __bf16 v8bf;
typedef __attribute__((ext_vector_type(8)))  float  v8f;

#define T_STEPS 256
#define BATCH   64
#define IN      1024
#define HID     1024
#define G4      4096   // 4*HID

// ---------------------------------------------------------------------------
// Init: convert weights to bf16, fuse biases, seed c / h(bf16) from c0 / h0.
// ---------------------------------------------------------------------------
__global__ void init_kernel(const float* __restrict__ W_ih,
                            const float* __restrict__ W_hh,
                            const float* __restrict__ b_ih,
                            const float* __restrict__ b_hh,
                            const float* __restrict__ h0,
                            const float* __restrict__ c0,
                            __bf16* __restrict__ Wih_bf,
                            __bf16* __restrict__ Whh_bf,
                            float*  __restrict__ bias,
                            float*  __restrict__ c,
                            __bf16* __restrict__ hbuf0) {
  int idx = blockIdx.x * blockDim.x + threadIdx.x;
  int stride = gridDim.x * blockDim.x;
  const int NW = G4 * IN;                      // == G4*HID
  for (int i = idx; i < NW; i += stride) {
    Wih_bf[i] = (__bf16)W_ih[i];
    Whh_bf[i] = (__bf16)W_hh[i];
  }
  for (int i = idx; i < G4; i += stride) bias[i] = b_ih[i] + b_hh[i];
  for (int i = idx; i < BATCH * HID; i += stride) {
    c[i]     = c0[i];
    hbuf0[i] = (__bf16)h0[i];
  }
}

// ---------------------------------------------------------------------------
// Phase 1 GEMM: xg[m, g] = sum_k x[m,k] * W_ih[g,k] + bias[g]
//   M = T*B = 16384, N = 4096, K = 1024 (A row-major, B = W_ih rows = K-major)
// Block: 128 threads (4 waves). Block tile 64(M) x 64(N); wave w -> rows
// [16w,16w+16), 4 accumulators across N. K-step = 32 (bf16 WMMA).
// ---------------------------------------------------------------------------
__global__ __launch_bounds__(128)
void gemm_xg_kernel(const float*  __restrict__ x,
                    const __bf16* __restrict__ Wih,
                    const float*  __restrict__ bias,
                    float*        __restrict__ xg) {
  __shared__ __bf16 lA[64][40];   // [m][k] tile (+pad)
  __shared__ __bf16 lB[64][40];   // [n][k] tile (+pad)

  const int tid  = threadIdx.x;
  const int wave = tid >> 5;
  const int lane = tid & 31;
  const int m0   = blockIdx.y * 64;
  const int n0   = blockIdx.x * 64;

  v8f acc[4] = {};

  for (int k0 = 0; k0 < IN; k0 += 32) {
    // stage A (convert f32 -> bf16) and B (already bf16) into LDS
    for (int i = tid; i < 64 * 32; i += 128) {
      int r = i >> 5, cc = i & 31;
      lA[r][cc] = (__bf16)x[(size_t)(m0 + r) * IN + k0 + cc];
    }
    for (int i = tid; i < 64 * 32; i += 128) {
      int r = i >> 5, cc = i & 31;
      lB[r][cc] = Wih[(size_t)(n0 + r) * IN + k0 + cc];
    }
    __syncthreads();

    // A fragment: 16x32 bf16; lane<16 -> K 0..15 ; lane>=16 -> K 16..31
    v16bf a;
    const int am = wave * 16 + (lane & 15);
    const int ko = (lane >> 4) * 16;
    #pragma unroll
    for (int e = 0; e < 16; ++e) {
      int kk = e + ko;
      a[e] = lA[am][kk];
    }

    #pragma unroll
    for (int nt = 0; nt < 4; ++nt) {
      // B fragment: 32x16 bf16; lane = column, lanes>=16 hold K 16..31
      v16bf b;
      const int bn = nt * 16 + (lane & 15);
      const int bk = (lane >> 4) * 16;
      #pragma unroll
      for (int e = 0; e < 16; ++e) b[e] = lB[bn][bk + e];
      acc[nt] = __builtin_amdgcn_wmma_f32_16x16x32_bf16(
          false, a, false, b, (short)0, acc[nt], false, false);
    }
    __syncthreads();
  }

  // epilogue: C/D layout -> vgpr r, lane l : M = r + 8*(l/16), N = l%16
  const int col   = lane & 15;
  const int rbase = m0 + wave * 16 + ((lane >> 4) << 3);
  #pragma unroll
  for (int nt = 0; nt < 4; ++nt) {
    const int g  = n0 + nt * 16 + col;
    const float bv = bias[g];
    #pragma unroll
    for (int r = 0; r < 8; ++r) {
      xg[(size_t)(rbase + r) * G4 + g] = acc[nt][r] + bv;
    }
  }
}

// ---------------------------------------------------------------------------
// Phase 2 per-timestep kernel (latency critical, 256 serial launches).
// Block owns hidden cols [j0, j0+16) for ALL batch rows and ALL 4 gates.
// 4 waves; wave w -> batch rows [16w,16w+16), one 16x16 accumulator per gate.
// No LDS, no barriers: WMMA fragments are loaded straight from global memory
// (L2-resident: h is 128 KB, W_hh is 8 MB) as contiguous 16/32-byte chunks
// in one contiguous k order shared by A and B (lane half h holds K 16h..16h+15).
//   z[b, gate*H + j] = xg_t[...] + sum_k h[b,k] * W_hh[gate*H + j, k]
// ---------------------------------------------------------------------------
__global__ __launch_bounds__(128)
void step_kernel(const float*  __restrict__ xg_t,   // [BATCH, G4] (slice at t)
                 const __bf16* __restrict__ Whh,    // [G4, HID] bf16
                 const __bf16* __restrict__ h_in,   // [BATCH, HID] bf16
                 float*        __restrict__ c,      // [BATCH, HID] f32 (in/out)
                 __bf16*       __restrict__ h_out,  // [BATCH, HID] bf16
                 float*        __restrict__ y_t) {  // [BATCH, HID] f32
  const int tid  = threadIdx.x;
  const int wave = tid >> 5;
  const int lane = tid & 31;
  const int lrow = lane & 15;
  const int hi   = lane >> 4;          // 0: low half-lanes, 1: high half-lanes
  const int j0   = blockIdx.x * 16;

  // Per-lane row bases (A: this wave's batch row; B: one W_hh row per gate)
  const __bf16* arow  = h_in + (size_t)(wave * 16 + lrow) * HID;
  const __bf16* brow0 = Whh + (size_t)(0 * HID + j0 + lrow) * HID;
  const __bf16* brow1 = Whh + (size_t)(1 * HID + j0 + lrow) * HID;
  const __bf16* brow2 = Whh + (size_t)(2 * HID + j0 + lrow) * HID;
  const __bf16* brow3 = Whh + (size_t)(3 * HID + j0 + lrow) * HID;

  v8f acc[4] = {};

  #pragma unroll 2
  for (int k0 = 0; k0 < HID; k0 += 32) {
    // A fragment: lane<16 holds K 0..15, lane>=16 holds K 16..31 (contiguous)
    v8bf alo = *(const v8bf*)(arow + k0 + 16 * hi);
    v8bf ahi = *(const v8bf*)(arow + k0 + 16 * hi + 8);
    v16bf a = __builtin_shufflevector(alo, ahi, 0, 1, 2, 3, 4, 5, 6, 7,
                                      8, 9, 10, 11, 12, 13, 14, 15);
    // B fragment: lane = column (lrow), lanes>=16 hold K 16..31 (contiguous)
    v16bf b0 = *(const v16bf*)(brow0 + k0 + 16 * hi);
    v16bf b1 = *(const v16bf*)(brow1 + k0 + 16 * hi);
    v16bf b2 = *(const v16bf*)(brow2 + k0 + 16 * hi);
    v16bf b3 = *(const v16bf*)(brow3 + k0 + 16 * hi);

    acc[0] = __builtin_amdgcn_wmma_f32_16x16x32_bf16(
        false, a, false, b0, (short)0, acc[0], false, false);
    acc[1] = __builtin_amdgcn_wmma_f32_16x16x32_bf16(
        false, a, false, b1, (short)0, acc[1], false, false);
    acc[2] = __builtin_amdgcn_wmma_f32_16x16x32_bf16(
        false, a, false, b2, (short)0, acc[2], false, false);
    acc[3] = __builtin_amdgcn_wmma_f32_16x16x32_bf16(
        false, a, false, b3, (short)0, acc[3], false, false);
  }

  // epilogue: spike (Heaviside) + cell update; each (b,j) owned by one lane.
  const int rbase = wave * 16 + (hi << 3);
  const int j     = j0 + lrow;
  #pragma unroll
  for (int r = 0; r < 8; ++r) {
    const int    b  = rbase + r;
    const size_t bj = (size_t)b * HID + j;
    const size_t bg = (size_t)b * G4 + j;
    float zi = acc[0][r] + xg_t[bg + 0 * HID];
    float zf = acc[1][r] + xg_t[bg + 1 * HID];
    float zg = acc[2][r] + xg_t[bg + 2 * HID];
    float zo = acc[3][r] + xg_t[bg + 3 * HID];
    float si = (zi >= 0.f) ? 1.f : 0.f;
    float sf = (zf >= 0.f) ? 1.f : 0.f;
    float sg = (zg >= 0.f) ? 1.f : 0.f;
    float so = (zo >= 0.f) ? 1.f : 0.f;
    float cn = sf * c[bj] + si * sg;
    float hn = so * cn;
    c[bj]     = cn;
    h_out[bj] = (__bf16)hn;
    y_t[bj]   = hn;
  }
}

// ---------------------------------------------------------------------------
// Launch
// ---------------------------------------------------------------------------
extern "C" void kernel_launch(void* const* d_in, const int* in_sizes, int n_in,
                              void* d_out, int out_size, void* d_ws, size_t ws_size,
                              hipStream_t stream) {
  const float* x    = (const float*)d_in[0];
  const float* h0   = (const float*)d_in[1];
  const float* c0   = (const float*)d_in[2];
  const float* W_ih = (const float*)d_in[3];
  const float* b_ih = (const float*)d_in[4];
  const float* W_hh = (const float*)d_in[5];
  const float* b_hh = (const float*)d_in[6];
  float* y = (float*)d_out;

  char* ws = (char*)d_ws;
  size_t off = 0;
  auto wsalloc = [&](size_t bytes) -> void* {
    void* p = ws + off;
    off = (off + bytes + 255) & ~(size_t)255;
    return p;
  };
  float*  xg     = (float*) wsalloc((size_t)T_STEPS * BATCH * G4 * sizeof(float));
  __bf16* Wih_bf = (__bf16*)wsalloc((size_t)G4 * IN  * sizeof(__bf16));
  __bf16* Whh_bf = (__bf16*)wsalloc((size_t)G4 * HID * sizeof(__bf16));
  float*  bias   = (float*) wsalloc((size_t)G4 * sizeof(float));
  float*  cbuf   = (float*) wsalloc((size_t)BATCH * HID * sizeof(float));
  __bf16* hb0    = (__bf16*)wsalloc((size_t)BATCH * HID * sizeof(__bf16));
  __bf16* hb1    = (__bf16*)wsalloc((size_t)BATCH * HID * sizeof(__bf16));
  (void)ws_size; (void)in_sizes; (void)n_in; (void)out_size;

  init_kernel<<<1024, 256, 0, stream>>>(W_ih, W_hh, b_ih, b_hh, h0, c0,
                                        Wih_bf, Whh_bf, bias, cbuf, hb0);

  dim3 gridA(G4 / 64, (T_STEPS * BATCH) / 64);
  gemm_xg_kernel<<<gridA, 128, 0, stream>>>(x, Wih_bf, bias, xg);

  for (int t = 0; t < T_STEPS; ++t) {
    const __bf16* hin  = (t & 1) ? hb1 : hb0;
    __bf16*       hout = (t & 1) ? hb0 : hb1;
    step_kernel<<<HID / 16, 128, 0, stream>>>(
        xg + (size_t)t * BATCH * G4, Whh_bf, hin, cbuf, hout,
        y + (size_t)t * BATCH * HID);
  }
}
